// GlobalGNN_16363825397777
// MI455X (gfx1250) — matrix operand order repaired and checked by reading, not compiled, not run on hardware
//
#include <hip/hip_runtime.h>
#include <hip/hip_bf16.h>

// ---- problem constants (shapes fixed by reference) ----
#define GAT_D   96
#define GAT_H   4
#define GAT_HD  384       // H*D
#define GAT_L   2
#define NEG_SLOPE 0.2f

typedef __bf16 bf16_t;
typedef bf16_t v16bf __attribute__((ext_vector_type(16)));
typedef float  v8f   __attribute__((ext_vector_type(8)));
typedef unsigned u32x4 __attribute__((ext_vector_type(4)));

union FragBF { v16bf v; u32x4 q[2]; };

// ---------- helpers: order-preserving float<->uint encode for atomicMax ----------
__device__ __forceinline__ unsigned fenc(float x) {
    unsigned b = __float_as_uint(x);
    return (b & 0x80000000u) ? ~b : (b | 0x80000000u);
}
__device__ __forceinline__ float fdec(unsigned e) {
    return __uint_as_float((e & 0x80000000u) ? (e & 0x7fffffffu) : ~e);
}

// ---------- conversions ----------
__global__ void k_cvt_x(const float* __restrict__ x, bf16_t* __restrict__ xb, int total) {
    int i = blockIdx.x * blockDim.x + threadIdx.x;
    if (i < total) xb[i] = (bf16_t)x[i];
}

// W [L, D, HD] row-major  ->  Wt [L, HD, D] row-major bf16 (pre-transposed so
// B-fragment K runs contiguous in memory)
__global__ void k_cvt_w(const float* __restrict__ W, bf16_t* __restrict__ Wt) {
    int i = blockIdx.x * blockDim.x + threadIdx.x;
    if (i >= GAT_L * GAT_HD * GAT_D) return;
    int l = i / (GAT_HD * GAT_D);
    int r = i % (GAT_HD * GAT_D);
    int j = r / GAT_D;            // output column  (0..383)
    int k = r % GAT_D;            // input  dim     (0..95)
    Wt[i] = (bf16_t)W[(l * GAT_D + k) * GAT_HD + j];
}

// ---------- GEMM: h[N,384] = xb[N,96] * Wt^T  via bf16 WMMA, f32 accumulate ----------
// wave32: one wave per 16-row M tile; A panel (16x96) resident; loop 24 N tiles.
__global__ void k_gemm(const bf16_t* __restrict__ xb, const bf16_t* __restrict__ wt,
                       float* __restrict__ h, int n) {
    int wave  = threadIdx.x >> 5;
    int lane  = threadIdx.x & 31;
    int mtile = blockIdx.x * 8 + wave;
    int mtiles = n >> 4;                    // n divisible by 16 (50000 = 3125*16)
    if (mtile >= mtiles) return;
    int m0 = mtile << 4;

    // A fragments: lane L -> row m0+(L&15); K chunks {0..7,16..23} / {8..15,24..31}
    int arow = m0 + (lane & 15);
    if (arow >= n) arow = n - 1;
    int akb = (lane >> 4) << 3;             // 0 or 8
    const bf16_t* ap = xb + (size_t)arow * GAT_D;
    FragBF a[3];
#pragma unroll
    for (int kk = 0; kk < 3; ++kk) {
        int kb = kk * 32 + akb;
        a[kk].q[0] = *(const u32x4*)(ap + kb);        // K = kb..kb+7
        a[kk].q[1] = *(const u32x4*)(ap + kb + 16);   // K = kb+16..kb+23
    }

    // B fragments: lane L -> col n0+(L&15); K chunks {0..7,16..23} / {8..15,24..31}
    int bj = lane & 15;
    int bk = (lane >> 4) << 3;              // 0 or 8
#pragma unroll 1
    for (int nt = 0; nt < GAT_HD / 16; ++nt) {
        const bf16_t* bp = wt + (size_t)(nt * 16 + bj) * GAT_D + bk;
        v8f acc = {};
#pragma unroll
        for (int kk = 0; kk < 3; ++kk) {
            FragBF b;
            b.q[0] = *(const u32x4*)(bp + kk * 32);
            b.q[1] = *(const u32x4*)(bp + kk * 32 + 16);
            acc = __builtin_amdgcn_wmma_f32_16x16x32_bf16(
                      false, a[kk].v, false, b.v, (short)0, acc, false, false);
        }
        // C layout: VGPR r, lane L -> (M = (L>=16?8:0)+r, N = L&15)
        float* cp = h + (size_t)(m0 + ((lane >> 4) << 3)) * GAT_HD + nt * 16 + (lane & 15);
#pragma unroll
        for (int r = 0; r < 8; ++r) cp[(size_t)r * GAT_HD] = acc[r];
    }
}

// ---------- per-node attention logits: alpha_{src,dst}[n,h] ----------
__global__ void k_alpha(const float* __restrict__ h, const float* __restrict__ asrc,
                        const float* __restrict__ adst, float* __restrict__ al_s,
                        float* __restrict__ al_d, int n) {
    int i = blockIdx.x * blockDim.x + threadIdx.x;
    if (i >= n * GAT_H) return;
    int node = i >> 2, hh = i & 3;
    const float* hp = h + (size_t)node * GAT_HD + hh * GAT_D;
    const float* as = asrc + hh * GAT_D;
    const float* ad = adst + hh * GAT_D;
    float s = 0.f, t = 0.f;
#pragma unroll 4
    for (int d = 0; d < GAT_D; ++d) { float v = hp[d]; s += v * as[d]; t += v * ad[d]; }
    al_s[i] = s;
    al_d[i] = t;
}

// ---------- edge scores + segment max (per dst node) ----------
__global__ void k_score(const long long* __restrict__ ei, const float* __restrict__ al_s,
                        const float* __restrict__ al_d, float* __restrict__ sc,
                        unsigned* __restrict__ menc, int E, int Etot) {
    int e = blockIdx.x * blockDim.x + threadIdx.x;
    if (e >= Etot) return;
    int src, dst;
    if (e < E) { src = (int)ei[e]; dst = (int)ei[E + e]; }
    else       { src = e - E;      dst = src; }
#pragma unroll
    for (int hh = 0; hh < GAT_H; ++hh) {
        float v = al_s[src * GAT_H + hh] + al_d[dst * GAT_H + hh];
        v = v > 0.f ? v : NEG_SLOPE * v;
        sc[(size_t)e * GAT_H + hh] = v;
        atomicMax(&menc[dst * GAT_H + hh], fenc(v));
    }
}

// ---------- exp(e - max) + segment sum of p ----------
__global__ void k_exp(const long long* __restrict__ ei, float* __restrict__ sc,
                      const unsigned* __restrict__ menc, float* __restrict__ denom,
                      int E, int Etot) {
    int e = blockIdx.x * blockDim.x + threadIdx.x;
    if (e >= Etot) return;
    int dst = (e < E) ? (int)ei[E + e] : (e - E);
#pragma unroll
    for (int hh = 0; hh < GAT_H; ++hh) {
        float p = __expf(sc[(size_t)e * GAT_H + hh] - fdec(menc[dst * GAT_H + hh]));
        sc[(size_t)e * GAT_H + hh] = p;
        atomicAdd(&denom[dst * GAT_H + hh], p);
    }
}

// ---------- weighted message aggregation: out[dst] += h[src] * alpha ----------
// one wave per edge; 32 lanes sweep the 384-wide feature (12 iterations)
__global__ void k_agg(const long long* __restrict__ ei, const float* __restrict__ h,
                      const float* __restrict__ sc, const float* __restrict__ denom,
                      float* __restrict__ outh, int E, int Etot) {
    int e = blockIdx.x * 8 + (threadIdx.x >> 5);
    if (e >= Etot) return;
    int lane = threadIdx.x & 31;
    int src, dst;
    if (e < E) { src = (int)ei[e]; dst = (int)ei[E + e]; }
    else       { src = e - E;      dst = src; }
    float a4[GAT_H];
#pragma unroll
    for (int hh = 0; hh < GAT_H; ++hh)
        a4[hh] = sc[(size_t)e * GAT_H + hh] / (denom[dst * GAT_H + hh] + 1e-16f);
    const float* hp = h + (size_t)src * GAT_HD;
    float* op = outh + (size_t)dst * GAT_HD;
#pragma unroll
    for (int i = 0; i < GAT_HD / 32; ++i) {
        int d = lane + 32 * i;
        atomicAdd(&op[d], hp[d] * a4[d / GAT_D]);
    }
}

// ---------- head mean + bias; partial BN stats ----------
__global__ void k_mean(const float* __restrict__ outh, const float* __restrict__ bias,
                       float* __restrict__ outd, float* __restrict__ sums, int n) {
    int d = threadIdx.x;                          // 96 threads
    int chunk = (n + gridDim.x - 1) / gridDim.x;
    int n0 = blockIdx.x * chunk;
    int n1 = n0 + chunk; if (n1 > n) n1 = n;
    float b = bias[d], s = 0.f, ss = 0.f;
    for (int node = n0; node < n1; ++node) {
        const float* p = outh + (size_t)node * GAT_HD;
        float v = (p[d] + p[GAT_D + d] + p[2 * GAT_D + d] + p[3 * GAT_D + d]) * 0.25f + b;
        outd[(size_t)node * GAT_D + d] = v;
        s += v; ss += v * v;
    }
    atomicAdd(&sums[d], s);
    atomicAdd(&sums[GAT_D + d], ss);
}

// ---------- finalize BN stats -> scale/shift ----------
__global__ void k_stats(const float* __restrict__ sums, const float* __restrict__ gamma,
                        const float* __restrict__ beta, float* __restrict__ sshift, int n) {
    int d = threadIdx.x;
    if (d >= GAT_D) return;
    float inv_n = 1.0f / (float)n;
    float mu  = sums[d] * inv_n;
    float var = sums[GAT_D + d] * inv_n - mu * mu;
    float sc  = rsqrtf(var + 1e-5f) * gamma[d];
    sshift[d]          = sc;
    sshift[GAT_D + d]  = beta[d] - mu * sc;
}

// ---------- normalize + relu + residual; also emit next-layer bf16 activations ----------
__global__ void k_finish(const float* __restrict__ outd, const float* __restrict__ sshift,
                         const float* __restrict__ xres, float* __restrict__ xout,
                         bf16_t* __restrict__ xbout, int total) {
    int i = blockIdx.x * blockDim.x + threadIdx.x;
    if (i >= total) return;
    int d = i % GAT_D;
    float v = outd[i] * sshift[d] + sshift[GAT_D + d];
    v = v > 0.f ? v : 0.f;
    v += xres[i];
    xout[i] = v;
    if (xbout) xbout[i] = (bf16_t)v;
}

extern "C" void kernel_launch(void* const* d_in, const int* in_sizes, int n_in,
                              void* d_out, int out_size, void* d_ws, size_t ws_size,
                              hipStream_t stream) {
    (void)n_in; (void)out_size; (void)ws_size;
    const float*     x     = (const float*)d_in[0];
    const long long* ei    = (const long long*)d_in[1];   // int64 [2,E]
    const float*     W     = (const float*)d_in[2];       // [L,D,HD]
    const float*     asrc  = (const float*)d_in[3];       // [L,H,D]
    const float*     adst  = (const float*)d_in[4];
    const float*     bias  = (const float*)d_in[5];       // [L,D]
    const float*     gamma = (const float*)d_in[6];
    const float*     beta  = (const float*)d_in[7];
    float*           out   = (float*)d_out;               // [N,D] f32

    const int n    = in_sizes[0] / GAT_D;
    const int E    = in_sizes[1] / 2;
    const int Etot = E + n;

    // ---- carve workspace ----
    char* ws = (char*)d_ws;
    size_t off = 0;
    auto alloc = [&](size_t bytes) -> void* {
        void* p = ws + off;
        off += (bytes + 255) & ~(size_t)255;
        return p;
    };
    bf16_t*   xb     = (bf16_t*)  alloc((size_t)n * GAT_D * 2);
    bf16_t*   wt     = (bf16_t*)  alloc((size_t)GAT_L * GAT_HD * GAT_D * 2);
    float*    h      = (float*)   alloc((size_t)n * GAT_HD * 4);
    float*    outh   = (float*)   alloc((size_t)n * GAT_HD * 4);
    float*    outd   = (float*)   alloc((size_t)n * GAT_D * 4);
    float*    x1     = (float*)   alloc((size_t)n * GAT_D * 4);
    float*    al_s   = (float*)   alloc((size_t)n * GAT_H * 4);
    float*    al_d   = (float*)   alloc((size_t)n * GAT_H * 4);
    unsigned* menc   = (unsigned*)alloc((size_t)n * GAT_H * 4);
    float*    denom  = (float*)   alloc((size_t)n * GAT_H * 4);
    float*    sc     = (float*)   alloc((size_t)Etot * GAT_H * 4);
    float*    sums   = (float*)   alloc(2 * GAT_D * 4);
    float*    sshift = (float*)   alloc(2 * GAT_D * 4);

    // ---- one-time conversions ----
    k_cvt_x<<<(n * GAT_D + 255) / 256, 256, 0, stream>>>(x, xb, n * GAT_D);
    k_cvt_w<<<(GAT_L * GAT_HD * GAT_D + 255) / 256, 256, 0, stream>>>(W, wt);

    for (int l = 0; l < GAT_L; ++l) {
        const float* xin  = (l == 0) ? x  : x1;
        float*       xout = (l == 0) ? x1 : out;

        hipMemsetAsync(menc,  0, (size_t)n * GAT_H * 4, stream);
        hipMemsetAsync(denom, 0, (size_t)n * GAT_H * 4, stream);
        hipMemsetAsync(outh,  0, (size_t)n * GAT_HD * 4, stream);
        hipMemsetAsync(sums,  0, 2 * GAT_D * 4, stream);

        k_gemm<<<(n / 16 + 7) / 8, 256, 0, stream>>>(xb, wt + (size_t)l * GAT_HD * GAT_D, h, n);
        k_alpha<<<(n * GAT_H + 127) / 128, 128, 0, stream>>>(
            h, asrc + l * GAT_H * GAT_D, adst + l * GAT_H * GAT_D, al_s, al_d, n);
        k_score<<<(Etot + 255) / 256, 256, 0, stream>>>(ei, al_s, al_d, sc, menc, E, Etot);
        k_exp  <<<(Etot + 255) / 256, 256, 0, stream>>>(ei, sc, menc, denom, E, Etot);
        k_agg  <<<(Etot + 7) / 8,     256, 0, stream>>>(ei, h, sc, denom, outh, E, Etot);
        k_mean <<<256, GAT_D, 0, stream>>>(outh, bias + l * GAT_D, outd, sums, n);
        k_stats<<<1, 128, 0, stream>>>(sums, gamma + l * GAT_D, beta + l * GAT_D, sshift, n);
        k_finish<<<(n * GAT_D + 255) / 256, 256, 0, stream>>>(
            outd, sshift, xin, xout, (l == 0) ? xb : (bf16_t*)nullptr, n * GAT_D);
    }
}
